// ClusterLayer_34969623724584
// MI455X (gfx1250) — matrix operand order repaired and checked by reading, not compiled, not run on hardware
//
#include <hip/hip_runtime.h>

typedef __attribute__((ext_vector_type(16))) _Float16 v16h;
typedef __attribute__((ext_vector_type(8)))  _Float16 f16x8;
typedef __attribute__((ext_vector_type(4)))  _Float16 f16x4;
typedef __attribute__((ext_vector_type(8)))  float    v8f;

#define IN_DIM   1024
#define BATCH    4096
#define N_OUT    8000
#define CLUSTER  10

#define M_TILE   80       // 8 whole clusters per tile
#define N_TILE   128
#define K_CHUNK  64
#define A_STRIDE 72       // halves (64 data + 8 pad) -> 144B rows, conflict-free b128 reads
#define B_STRIDE 72
#define A_BYTES  (M_TILE * A_STRIDE * 2)          // 11520
#define B_BYTES  (N_TILE * B_STRIDE * 2)          // 18432
#define DUMP_STRIDE 19                            // floats; keeps lane-halves / waves on disjoint banks
#define DUMP_BYTES  (8 * M_TILE * DUMP_STRIDE * 4) // 48640
#define SMEM_BYTES  ((A_BYTES + B_BYTES) > DUMP_BYTES ? (A_BYTES + B_BYTES) : DUMP_BYTES)

__global__ __launch_bounds__(256)
void cluster_wta_gemm(const float* __restrict__ inp,  // [IN_DIM][BATCH]
                      const float* __restrict__ ker,  // [N_OUT][IN_DIM]
                      float* __restrict__ out)        // [N_OUT][BATCH]
{
    __shared__ __align__(16) unsigned char smem[SMEM_BYTES];
    _Float16* Alds = (_Float16*)smem;                 // [80][72] halves, row-major in K
    _Float16* Blds = (_Float16*)(smem + A_BYTES);     // [128][72] halves, [n][k]
    float*    dump = (float*)smem;                    // reused after GEMM

    const int tid   = threadIdx.x;
    const int lane  = tid & 31;
    const int wave  = tid >> 5;
    const int lm    = lane & 15;      // 0..15
    const int khi   = lane >> 4;      // lane-half selects K sub-range
    const int mBase = blockIdx.y * M_TILE;
    const int nBase = blockIdx.x * N_TILE;

    v8f acc[5];
    const v8f vzero = {0.f, 0.f, 0.f, 0.f, 0.f, 0.f, 0.f, 0.f};
#pragma unroll
    for (int t = 0; t < 5; ++t) acc[t] = vzero;

    for (int kc = 0; kc < IN_DIM; kc += K_CHUNK) {
        // ---- stage A tile: 80x64 f32 -> f16 into Alds[r][k] (1280 float4, 5/thread) ----
#pragma unroll
        for (int i = 0; i < 5; ++i) {
            int idx = tid + 256 * i;
            int r   = idx >> 4;            // 16 float4 per row
            int c4  = idx & 15;
            const float4 v = *((const float4*)(ker + (size_t)(mBase + r) * IN_DIM + kc) + c4);
            f16x4 h = { (_Float16)v.x, (_Float16)v.y, (_Float16)v.z, (_Float16)v.w };
            *(f16x4*)(Alds + r * A_STRIDE + (c4 << 2)) = h;
        }
        // ---- stage B tile: 64x128 f32 -> f16, transposed into Blds[n][k] ----
        // 4x4 register transpose blocks: 16 row-groups x 32 col-groups = 512 tasks, 2/thread
#pragma unroll
        for (int i = 0; i < 2; ++i) {
            int task = tid + 256 * i;
            int rg   = task >> 5;          // 0..15 (groups of 4 k-rows)
            int cg   = task & 31;          // 0..31 (groups of 4 cols)
            const float* gp = inp + (size_t)(kc + rg * 4) * BATCH + nBase + cg * 4;
            float4 r0 = *(const float4*)(gp);
            float4 r1 = *(const float4*)(gp + BATCH);
            float4 r2 = *(const float4*)(gp + 2 * BATCH);
            float4 r3 = *(const float4*)(gp + 3 * BATCH);
            _Float16* bp = Blds + (cg * 4) * B_STRIDE + rg * 4;
            f16x4 c0 = {(_Float16)r0.x, (_Float16)r1.x, (_Float16)r2.x, (_Float16)r3.x};
            f16x4 c1 = {(_Float16)r0.y, (_Float16)r1.y, (_Float16)r2.y, (_Float16)r3.y};
            f16x4 c2 = {(_Float16)r0.z, (_Float16)r1.z, (_Float16)r2.z, (_Float16)r3.z};
            f16x4 c3 = {(_Float16)r0.w, (_Float16)r1.w, (_Float16)r2.w, (_Float16)r3.w};
            *(f16x4*)(bp)                = c0;
            *(f16x4*)(bp + B_STRIDE)     = c1;
            *(f16x4*)(bp + 2 * B_STRIDE) = c2;
            *(f16x4*)(bp + 3 * B_STRIDE) = c3;
        }
        __syncthreads();

        // ---- WMMA: two K=32 steps per chunk, 5 M-subtiles per wave ----
#pragma unroll
        for (int ks = 0; ks < K_CHUNK; ks += 32) {
            // B fragment (32x16): halves 0-7 -> K = ks+8*khi.., halves 8-15 -> +16
            const _Float16* bp = Blds + (wave * 16 + lm) * B_STRIDE + ks + khi * 8;
            f16x8 b0 = *(const f16x8*)(bp);
            f16x8 b1 = *(const f16x8*)(bp + 16);
            v16h bf;
#pragma unroll
            for (int e = 0; e < 8; ++e) { bf[e] = b0[e]; bf[e + 8] = b1[e]; }
#pragma unroll
            for (int t = 0; t < 5; ++t) {
                // A fragment (16x32): halves 0-7 -> K = ks+8*khi.., halves 8-15 -> +16
                const _Float16* ap = Alds + (t * 16 + lm) * A_STRIDE + ks + khi * 8;
                f16x8 a0 = *(const f16x8*)(ap);
                f16x8 a1 = *(const f16x8*)(ap + 16);
                v16h af;
#pragma unroll
                for (int e = 0; e < 8; ++e) { af[e] = a0[e]; af[e + 8] = a1[e]; }
                acc[t] = __builtin_amdgcn_wmma_f32_16x16x32_f16(
                    false, af, false, bf, (short)0, acc[t], false, false);
            }
        }
        __syncthreads();
    }

    // ---- dump accumulators: wave-private [80][19] f32 region ----
    float* dreg = dump + wave * (M_TILE * DUMP_STRIDE);
#pragma unroll
    for (int t = 0; t < 5; ++t) {
#pragma unroll
        for (int v = 0; v < 8; ++v) {
            int row = t * 16 + v + khi * 8;   // C/D layout: VGPR v -> M = v + 8*lanehalf
            dreg[row * DUMP_STRIDE + lm] = acc[t][v];
        }
    }
    __syncthreads();

    // ---- segmented winner-take-all: 8 clusters x 128 cols = 1024 tasks, 4/thread ----
#pragma unroll
    for (int i = 0; i < 4; ++i) {
        int task = tid + 256 * i;
        int cl   = task >> 7;       // 0..7
        int col  = task & 127;      // 0..127
        int w    = col >> 4;
        int lc   = col & 15;
        const float* p = dump + w * (M_TILE * DUMP_STRIDE) + (cl * CLUSTER) * DUMP_STRIDE + lc;
        float best = p[0];
        int   bi   = 0;
#pragma unroll
        for (int j = 1; j < CLUSTER; ++j) {
            float v = p[j * DUMP_STRIDE];
            if (v > best) { best = v; bi = j; }   // strict '>' => first max wins (jnp.argmax)
        }
        float* op = out + (size_t)(mBase + cl * CLUSTER) * BATCH + nBase + col;
#pragma unroll
        for (int j = 0; j < CLUSTER; ++j) {
            op[(size_t)j * BATCH] = (j == bi) ? 1.0f : 0.0f;
        }
    }
}

extern "C" void kernel_launch(void* const* d_in, const int* in_sizes, int n_in,
                              void* d_out, int out_size, void* d_ws, size_t ws_size,
                              hipStream_t stream) {
    (void)in_sizes; (void)n_in; (void)d_ws; (void)ws_size; (void)out_size;
    const float* inp = (const float*)d_in[0];   // [1024, 4096]
    const float* ker = (const float*)d_in[1];   // [8000, 1024]
    float* out = (float*)d_out;                 // [8000, 4096]

    dim3 grid(BATCH / N_TILE, N_OUT / M_TILE);  // (32, 100)
    dim3 block(256);
    cluster_wta_gemm<<<grid, block, 0, stream>>>(inp, ker, out);
}
